// WorldModelLoss_77996606095989
// MI455X (gfx1250) — matrix operand order repaired and checked by reading, not compiled, not run on hardware
//
#include <hip/hip_runtime.h>
#include <hip/hip_bf16.h>

typedef __attribute__((ext_vector_type(16))) _Float16 v16h;
typedef __attribute__((ext_vector_type(8)))  float    v8f;

// SSIM constants
#define C1F 1.0e-4f      // 0.01^2
#define C2F 9.0e-4f      // 0.03^2

// Problem shape (fixed by setup_inputs): N=64, F=4, H=128, W=128, C=3
// SSIM planes = N * F*C = 768, each 128x128; strips of 16 rows -> 8 per plane.
constexpr int NBLK_MSE  = 1024;
constexpr int NBLK_KL   = 128;
constexpr int NBLK_SSIM = 768 * 8;   // 6144

// ws layout (floats)
constexpr int OFF_RECON = 0;
constexpr int OFF_PRED  = OFF_RECON + NBLK_MSE;   // 1024
constexpr int OFF_KL    = OFF_PRED  + NBLK_MSE;   // 2048
constexpr int OFF_DYN   = OFF_KL    + NBLK_KL;    // 2176
constexpr int OFF_HUB   = OFF_DYN   + NBLK_KL;    // 2304
constexpr int OFF_SSIM  = OFF_HUB   + 1;          // 2305

// Normalized 11-tap Gaussian (sigma=1.5) band weight, computed branchlessly:
// g[d] = exp(-(d-5)^2/4.5) / 3.7592318  for d in [0,10], else 0.
__device__ inline _Float16 gaussw(int d) {
    float t = (float)(d - 5);
    float v = __expf(t * t * (-1.f / 4.5f)) * 0.26601172f;
    return (_Float16)(((d >= 0) && (d <= 10)) ? v : 0.f);
}

// ---------------------------------------------------------------------------
// Streaming MSE partial sums (float4 vectorized, deterministic tree reduce)
// ---------------------------------------------------------------------------
__global__ __launch_bounds__(256)
void mse_partial_kernel(const float4* __restrict__ a, const float4* __restrict__ b,
                        float* __restrict__ partial, int n4) {
    __shared__ float red[256];
    const int tid = threadIdx.x;
    float s = 0.f;
    for (int i = blockIdx.x * 256 + tid; i < n4; i += gridDim.x * 256) {
        float4 av = a[i], bv = b[i];
        float dx = av.x - bv.x, dy = av.y - bv.y;
        float dz = av.z - bv.z, dw = av.w - bv.w;
        s += dx*dx + dy*dy + dz*dz + dw*dw;
    }
    red[tid] = s; __syncthreads();
    for (int h = 128; h > 0; h >>= 1) {
        if (tid < h) red[tid] += red[tid + h];
        __syncthreads();
    }
    if (tid == 0) partial[blockIdx.x] = red[0];
}

// ---------------------------------------------------------------------------
// VAE KL: sum of (1 + logvar - mu^2 - exp(logvar))
// ---------------------------------------------------------------------------
__global__ __launch_bounds__(256)
void kl_partial_kernel(const float* __restrict__ mu, const float* __restrict__ lv,
                       float* __restrict__ partial, int n) {
    __shared__ float red[256];
    const int tid = threadIdx.x;
    float s = 0.f;
    for (int i = blockIdx.x * 256 + tid; i < n; i += gridDim.x * 256) {
        float l = lv[i], m = mu[i];
        s += 1.f + l - m * m - expf(l);
    }
    red[tid] = s; __syncthreads();
    for (int h = 128; h > 0; h >>= 1) {
        if (tid < h) red[tid] += red[tid + h];
        __syncthreads();
    }
    if (tid == 0) partial[blockIdx.x] = red[0];
}

// ---------------------------------------------------------------------------
// Gaussian-to-Gaussian KL (dynamics loss), clipped
// ---------------------------------------------------------------------------
__global__ __launch_bounds__(256)
void dyn_partial_kernel(const float* __restrict__ m1, const float* __restrict__ lv1,
                        const float* __restrict__ m2, const float* __restrict__ lv2,
                        float* __restrict__ partial, int n) {
    __shared__ float red[256];
    const int tid = threadIdx.x;
    float s = 0.f;
    for (int i = blockIdx.x * 256 + tid; i < n; i += gridDim.x * 256) {
        float l1 = fminf(fmaxf(lv1[i], -10.f), 10.f);
        float l2 = fminf(fmaxf(lv2[i], -10.f), 10.f);
        float dm = m2[i] - m1[i];
        float kl = 0.5f * (l2 - l1 + expf(l1 - l2) + dm * dm * expf(-l2) - 1.f);
        s += fminf(fmaxf(kl, 0.f), 100.f);
    }
    red[tid] = s; __syncthreads();
    for (int h = 128; h > 0; h >>= 1) {
        if (tid < h) red[tid] += red[tid + h];
        __syncthreads();
    }
    if (tid == 0) partial[blockIdx.x] = red[0];
}

// ---------------------------------------------------------------------------
// Huber reward loss (tiny: one block)
// ---------------------------------------------------------------------------
__global__ __launch_bounds__(256)
void huber_kernel(const float* __restrict__ pred, const float* __restrict__ tgt,
                  float* __restrict__ partial, int n) {
    __shared__ float red[256];
    const int tid = threadIdx.x;
    float s = 0.f;
    for (int i = tid; i < n; i += 256) {
        float t = fminf(fmaxf(tgt[i], -15.f), 15.f);
        float d = pred[i] - t;
        float ad = fabsf(d);
        s += (ad < 1.f) ? 0.5f * d * d : ad - 0.5f;
    }
    red[tid] = s; __syncthreads();
    for (int h = 128; h > 0; h >>= 1) {
        if (tid < h) red[tid] += red[tid + h];
        __syncthreads();
    }
    if (tid == 0) partial[0] = red[0];
}

// ---------------------------------------------------------------------------
// SSIM: one workgroup (128 thr = 4 waves) per 16x128 strip of one plane.
//
// Both separable conv passes run on the matrix pipe as banded matmuls:
//   vertical:   D(16x16) = Gband(16x32) x In(32x16)   [A=const, B from global]
//   horizontal: D(16x16) = Vt(16x32)    x Gband(32x16)[A from LDS, B=const]
// Vertical B-fragments are built straight from global loads into registers
// (products {x^2,y^2,xy} formed in flight); only the vertical-conv result
// goes through LDS (needed for the cross-tile column halo). 20 WMMAs/wave.
// ---------------------------------------------------------------------------
__global__ __launch_bounds__(128)
void ssim_partial_kernel(const float* __restrict__ x, const float* __restrict__ y,
                         float* __restrict__ partial) {
    // stride 168 halves: 16B-aligned 8-element runs + conflict-free 16-lane rows
    __shared__ _Float16 Vpad[5][16][168];  // vertical-conv out + zero col halos
    __shared__ float red[128];

    const int tid   = threadIdx.x;
    const int strip = blockIdx.x & 7;
    const int plane = blockIdx.x >> 3;
    const int n  = plane / 12;
    const int fc = plane % 12;
    const int f  = fc / 3;
    const int c  = fc % 3;
    const size_t baseNF = (size_t)(n * 4 + f) * (size_t)(128 * 128 * 3) + (size_t)c;
    const int r0 = strip * 16;

    // Zero Vpad halo columns [0,5) and [133,168) for all 5 signals x 16 rows
    for (int i = tid; i < 5 * 16 * 40; i += 128) {
        int hcol = i % 40;
        int row  = i / 40;                       // 0..79
        int col  = hcol < 5 ? hcol : 128 + hcol; // 0..4, 133..167
        Vpad[row >> 4][row & 15][col] = (_Float16)0.f;
    }

    const int lane = tid & 31;
    const int wv   = tid >> 5;     // wave id 0..3
    const int hlf  = lane >> 4;    // lane half
    const int nn   = lane & 15;

    // Constant Gaussian band fragments (branchless, no memory):
    // Vertical A(16x32):  A[m][k] = g[k-m]   (m = nn; 16-bit A layout)
    v16h aband;
#pragma unroll
    for (int e = 0; e < 16; ++e) {
        int k = e + (hlf ? 16 : 0);
        aband[e] = gaussw(k - nn);
    }
    // Horizontal B(32x16): B[k][n] = g[k-n]  (n = nn; 16-bit B layout)
    v16h bband;
#pragma unroll
    for (int e = 0; e < 16; ++e) {
        int k = e + (hlf ? 16 : 0);
        bband[e] = gaussw(k - nn);
    }

    // Pass 1: vertical banded matmul. B-fragments gathered from global:
    // lane covers rows r0-5+k (k = e + 16*hlf) of column n0+nn.
    // Rows k>=26 are annihilated by the zero band in A, so only image-bounds
    // masking is needed. D (f32) -> f16 Vpad.
#pragma unroll
    for (int ti = 0; ti < 2; ++ti) {
        const int n0  = (wv * 2 + ti) * 16;
        const int col = n0 + nn;
        v16h bsig[5];
#pragma unroll
        for (int e = 0; e < 16; ++e) {
            int hrow = r0 + e + (hlf ? 16 : 0) - 5;
            int hcl  = hrow < 0 ? 0 : (hrow > 127 ? 127 : hrow);
            float m  = (hrow == hcl) ? 1.f : 0.f;
            size_t off = baseNF + (size_t)hcl * 384 + (size_t)col * 3;
            float xv = x[off] * m;
            float yv = y[off] * m;
            bsig[0][e] = (_Float16)xv;
            bsig[1][e] = (_Float16)yv;
            bsig[2][e] = (_Float16)(xv * xv);
            bsig[3][e] = (_Float16)(yv * yv);
            bsig[4][e] = (_Float16)(xv * yv);
        }
#pragma unroll
        for (int s5 = 0; s5 < 5; ++s5) {
            v8f cz = {};
            v8f d = __builtin_amdgcn_wmma_f32_16x16x32_f16(
                false, aband, false, bsig[s5], (short)0, cz, false, false);
#pragma unroll
            for (int e = 0; e < 8; ++e)
                Vpad[s5][e + 8 * hlf][5 + col] = (_Float16)d[e];
        }
    }
    __syncthreads();   // halo zeros + all tiles' vertical results visible

    // Pass 2: horizontal banded matmul + SSIM map
    float ssum = 0.f;
#pragma unroll
    for (int ti = 0; ti < 2; ++ti) {
        const int n0 = (wv * 2 + ti) * 16;
        v8f acc[5];
#pragma unroll
        for (int s5 = 0; s5 < 5; ++s5) {
            v16h a;
#pragma unroll
            for (int e = 0; e < 16; ++e) {
                int k = e + (hlf ? 16 : 0);
                a[e] = Vpad[s5][nn][n0 + k];
            }
            v8f cz = {};
            acc[s5] = __builtin_amdgcn_wmma_f32_16x16x32_f16(
                false, a, false, bband, (short)0, cz, false, false);
        }
#pragma unroll
        for (int e = 0; e < 8; ++e) {
            float mx = acc[0][e], my = acc[1][e];
            float mxs = mx * mx, mys = my * my, mxy = mx * my;
            float sx   = acc[2][e] - mxs;
            float sy   = acc[3][e] - mys;
            float sxyv = acc[4][e] - mxy;
            float num = (2.f * mxy + C1F) * (2.f * sxyv + C2F);
            float den = (mxs + mys + C1F) * (sx + sy + C2F);
            ssum += num / den;
        }
    }

    red[tid] = ssum;
    __syncthreads();
    for (int h = 64; h > 0; h >>= 1) {
        if (tid < h) red[tid] += red[tid + h];
        __syncthreads();
    }
    if (tid == 0) partial[blockIdx.x] = red[0];
}

// ---------------------------------------------------------------------------
// Final combine: deterministic sums of all partial regions, apply weights
// ---------------------------------------------------------------------------
__device__ float region_sum256(const float* __restrict__ p, int n,
                               float* red, int tid) {
    float s = 0.f;
    for (int i = tid; i < n; i += 256) s += p[i];
    red[tid] = s; __syncthreads();
    for (int h = 128; h > 0; h >>= 1) {
        if (tid < h) red[tid] += red[tid + h];
        __syncthreads();
    }
    float r = red[0];
    __syncthreads();   // safe reuse of red by next region
    return r;
}

__global__ __launch_bounds__(256)
void combine_kernel(const float* __restrict__ ws, float* __restrict__ out) {
    __shared__ float red[256];
    const int tid = threadIdx.x;
    float recon = region_sum256(ws + OFF_RECON, NBLK_MSE,  red, tid);
    float pred  = region_sum256(ws + OFF_PRED,  NBLK_MSE,  red, tid);
    float kls   = region_sum256(ws + OFF_KL,    NBLK_KL,   red, tid);
    float dyns  = region_sum256(ws + OFF_DYN,   NBLK_KL,   red, tid);
    float ssims = region_sum256(ws + OFF_SSIM,  NBLK_SSIM, red, tid);
    if (tid == 0) {
        const float NE = 12582912.f;   // 64*4*128*128*3 (also SSIM pixel count)
        const float NL = 32768.f;      // 64*512
        float recon_mse   = recon / NE;
        float pred_mse    = pred  / NE;
        float kl_loss     = -0.5f * kls / NL;
        float dyn_loss    = dyns / NL;
        float reward_loss = ws[OFF_HUB] / 64.f;
        float ssim_val    = ssims / NE;
        out[0] = recon_mse + pred_mse + 0.1f * kl_loss + 1.0f * dyn_loss
               + reward_loss - 0.1f * ssim_val;
    }
}

// ---------------------------------------------------------------------------
extern "C" void kernel_launch(void* const* d_in, const int* in_sizes, int n_in,
                              void* d_out, int out_size, void* d_ws, size_t ws_size,
                              hipStream_t stream) {
    (void)n_in; (void)out_size; (void)ws_size;
    const float* frames          = (const float*)d_in[0];
    const float* frames_next     = (const float*)d_in[1];
    const float* rewards         = (const float*)d_in[2];
    const float* z_mu            = (const float*)d_in[4];
    const float* z_logvar        = (const float*)d_in[5];
    const float* frame_recon     = (const float*)d_in[6];
    const float* z_next_mu       = (const float*)d_in[8];
    const float* z_next_logvar   = (const float*)d_in[9];
    const float* frame_next_pred = (const float*)d_in[10];
    const float* reward_pred     = (const float*)d_in[11];
    const float* z_target_mu     = (const float*)d_in[12];
    const float* z_target_logvar = (const float*)d_in[13];

    float* ws  = (float*)d_ws;
    float* out = (float*)d_out;

    const int nF = in_sizes[0];   // 12582912
    const int nL = in_sizes[4];   // 32768
    const int nR = in_sizes[2];   // 64

    mse_partial_kernel<<<NBLK_MSE, 256, 0, stream>>>(
        (const float4*)frame_recon, (const float4*)frames, ws + OFF_RECON, nF / 4);
    mse_partial_kernel<<<NBLK_MSE, 256, 0, stream>>>(
        (const float4*)frame_next_pred, (const float4*)frames_next, ws + OFF_PRED, nF / 4);
    kl_partial_kernel<<<NBLK_KL, 256, 0, stream>>>(z_mu, z_logvar, ws + OFF_KL, nL);
    dyn_partial_kernel<<<NBLK_KL, 256, 0, stream>>>(
        z_next_mu, z_next_logvar, z_target_mu, z_target_logvar, ws + OFF_DYN, nL);
    huber_kernel<<<1, 256, 0, stream>>>(reward_pred, rewards, ws + OFF_HUB, nR);
    ssim_partial_kernel<<<NBLK_SSIM, 128, 0, stream>>>(
        frame_next_pred, frames_next, ws + OFF_SSIM);
    combine_kernel<<<1, 256, 0, stream>>>(ws, out);
}
